// PixelNeRFDensityField_3375844295144
// MI455X (gfx1250) — compile-verified, matrix operand order checked by reading, not run on hardware
//
#include <hip/hip_runtime.h>
#include <hip/hip_bf16.h>
#include <math.h>

// ---------------------------------------------------------------------------
// PixelNeRF density field, fused WMMA implementation for gfx1250 (MI455X).
//   M = 2*1024*48 = 98304 points, hidden = 512, 5 residual blocks.
//   prep : projection + bilinear gather + positional encoding -> f16
//   pack : weights -> WMMA B-fragment-major f16 layout
//   mlp  : 32-row tiles, 8 waves; each wave = 2 M-strips x 4 N-tiles so every
//          B-fragment load feeds two v_wmma_f32_16x16x32_f16 (halves L2
//          weight traffic vs 16-row tiles, which is the binding resource).
// ---------------------------------------------------------------------------

typedef _Float16 v4h  __attribute__((ext_vector_type(4)));
typedef _Float16 v8h  __attribute__((ext_vector_type(8)));
typedef _Float16 v16h __attribute__((ext_vector_type(16)));
typedef float    v8f  __attribute__((ext_vector_type(8)));

#define M_TOTAL   98304
#define RS_PER_B  49152
#define DH        512
#define DHP       520          // padded LDS stride (halfs), 16B-aligned rows
#define DINP      64           // 63 real + 1 pad
#define NB        5
#define MT        32           // rows per workgroup tile
#define FRAG_HALFS 512         // one 32x16 f16 B-fragment = 512 halfs
#define FRAGS_512  (32 * 16)   // (N/16)*(K/32) fragments for a 512x512 matrix
#define BLK_MAT_HALFS ((size_t)FRAGS_512 * FRAG_HALFS)  // 262144 halfs

// dynamic LDS: sH | sT | sZ | sE
#define SMEM_HALFS (3 * MT * DHP + MT * 72)
#define SMEM_BYTES (SMEM_HALFS * 2)

__device__ __forceinline__ v8f vzero8() { v8f z = {}; return z; }

// ---------------------------------------------------------------------------
// Kernel 1: projection, bilinear feature gather, positional encoding
// ---------------------------------------------------------------------------
__global__ __launch_bounds__(256) void nerf_prep_kernel(
    const float* __restrict__ xyz, const float* __restrict__ c2w,
    const float* __restrict__ kmat, const float* __restrict__ feat,
    _Float16* __restrict__ encW, _Float16* __restrict__ algW)
{
    const long m = blockIdx.x;
    const int  b = (m >= RS_PER_B) ? 1 : 0;

    __shared__ float sCam[3];
    __shared__ float sW[2];
    __shared__ int   sI[4];

    if (threadIdx.x == 0) {
        const float p0 = xyz[m * 3 + 0], p1 = xyz[m * 3 + 1], p2 = xyz[m * 3 + 2];
        const float* C = c2w + b * 16;
        const float q0 = p0 - C[3], q1 = p1 - C[7], q2 = p2 - C[11];
        float cam0 = C[0] * q0 + C[4] * q1 + C[8]  * q2;
        float cam1 = C[1] * q0 + C[5] * q1 + C[9]  * q2;
        float cam2 = C[2] * q0 + C[6] * q1 + C[10] * q2;
        const float* K = kmat + b * 9;
        float u = K[0] * cam0 + K[1] * cam1 + K[2] * cam2;
        float v = K[3] * cam0 + K[4] * cam1 + K[5] * cam2;
        float z = K[6] * cam0 + K[7] * cam1 + K[8] * cam2;
        if (fabsf(z) < 1e-6f) z = 1e-6f;
        float x = fminf(fmaxf(u / z, 0.f), 127.f);
        float y = fminf(fmaxf(v / z, 0.f), 127.f);
        float x0 = floorf(x), y0 = floorf(y);
        int x0i = (int)x0, y0i = (int)y0;
        sI[0] = x0i; sI[1] = y0i;
        sI[2] = (x0i + 1 < 127) ? x0i + 1 : 127;
        sI[3] = (y0i + 1 < 127) ? y0i + 1 : 127;
        sW[0] = x - x0; sW[1] = y - y0;
        sCam[0] = cam0; sCam[1] = cam1; sCam[2] = cam2;
    }
    __syncthreads();

    const float wx = sW[0], wy = sW[1];
    const int x0i = sI[0], y0i = sI[1], x1i = sI[2], y1i = sI[3];
    const float w00 = (1.f - wx) * (1.f - wy);
    const float w01 = wx * (1.f - wy);
    const float w10 = (1.f - wx) * wy;
    const float w11 = wx * wy;

    const float* fb = feat + (size_t)b * DH * 16384;
    for (int c = threadIdx.x; c < DH; c += 256) {
        const float* fc = fb + (size_t)c * 16384;
        float f00 = fc[y0i * 128 + x0i];
        float f01 = fc[y0i * 128 + x1i];
        float f10 = fc[y1i * 128 + x0i];
        float f11 = fc[y1i * 128 + x1i];
        algW[m * DH + c] = (_Float16)(f00 * w00 + f01 * w01 + f10 * w10 + f11 * w11);
    }

    if (threadIdx.x < DINP) {
        const int t = threadIdx.x;
        const float twopi = 6.28318530717958647692f;
        float val;
        if (t < 3) {
            val = sCam[t];
        } else if (t < 33) {
            int idx = t - 3; int j = idx / 10; int f = idx % 10;
            val = sinf(twopi * sCam[j] * (float)(1 << f));
        } else if (t < 63) {
            int idx = t - 33; int j = idx / 10; int f = idx % 10;
            val = cosf(twopi * sCam[j] * (float)(1 << f));
        } else {
            val = 0.f;
        }
        encW[m * DINP + t] = (_Float16)val;
    }
}

// ---------------------------------------------------------------------------
// Kernel 2: pack fp32 weights into WMMA 16-bit B-fragment layout.
//   k = k0 + (i<8 ? i : i+8) + (lane<16 ? 0 : 8),  n = n0 + (lane&15)
// stored lane-contiguous: dst[lane*16 + i]
// ---------------------------------------------------------------------------
__global__ __launch_bounds__(256) void nerf_pack_kernel(
    const float* __restrict__ win, const float* __restrict__ zw,
    const float* __restrict__ f0w, const float* __restrict__ f1w,
    const float* __restrict__ wout,
    _Float16* __restrict__ winP, _Float16* __restrict__ blkP,
    _Float16* __restrict__ woutP)
{
    const int NFRAG = 64 + 15 * FRAGS_512 + 16;
    const int fid  = blockIdx.x * 8 + (threadIdx.x >> 5);
    const int lane = threadIdx.x & 31;
    if (fid >= NFRAG) return;

    const int colN  = lane & 15;
    const int klane = (lane >> 4) << 3;

    if (fid < 64) {                        // lin_in_w : K=64(63 real), N=512
        const int nt = fid >> 1, kt = fid & 1;
        const int n0 = nt * 16, k0 = kt * 32;
        _Float16* dst = winP + (size_t)fid * FRAG_HALFS + lane * 16;
        #pragma unroll
        for (int i = 0; i < 16; ++i) {
            int k = k0 + ((i < 8) ? i : i + 8) + klane;
            float v = (k < 63) ? win[k * DH + (n0 + colN)] : 0.f;
            dst[i] = (_Float16)v;
        }
    } else if (fid < 64 + 15 * FRAGS_512) { // 5 x {Zw, F0, F1}, 512x512 each
        const int g  = fid - 64;
        const int mi = g >> 9;
        const int f  = g & 511;
        const int blk = mi / 3, which = mi % 3;
        const float* base =
            ((which == 0) ? zw : (which == 1) ? f0w : f1w) + (size_t)blk * DH * DH;
        const int nt = f >> 4, kt = f & 15;
        const int n0 = nt * 16, k0 = kt * 32;
        _Float16* dst = blkP + (size_t)g * FRAG_HALFS + lane * 16;
        #pragma unroll
        for (int i = 0; i < 16; ++i) {
            int k = k0 + ((i < 8) ? i : i + 8) + klane;
            dst[i] = (_Float16)base[k * DH + (n0 + colN)];
        }
    } else {                               // lin_out_w : K=512, N=16 (1 real)
        const int kt = fid - (64 + 15 * FRAGS_512);
        const int k0 = kt * 32;
        _Float16* dst = woutP + (size_t)kt * FRAG_HALFS + lane * 16;
        #pragma unroll
        for (int i = 0; i < 16; ++i) {
            int k = k0 + ((i < 8) ? i : i + 8) + klane;
            float v = (colN == 0) ? wout[k] : 0.f;
            dst[i] = (_Float16)v;
        }
    }
}

// ---------------------------------------------------------------------------
// Kernel 3: fused MLP, MT=32 rows per workgroup.
// ---------------------------------------------------------------------------
__device__ __forceinline__ v16h load_a_frag(const _Float16* A, int stride,
                                            int k0, int lane, bool relu)
{
    const int row = lane & 15;
    const int kb  = k0 + ((lane >> 4) << 3);
    const _Float16* p = A + row * stride + kb;
    v8h lo = *(const v8h*)(p);
    v8h hi = *(const v8h*)(p + 16);
    v16h a = __builtin_shufflevector(lo, hi, 0, 1, 2, 3, 4, 5, 6, 7,
                                     8, 9, 10, 11, 12, 13, 14, 15);
    if (relu) {
        #pragma unroll
        for (int i = 0; i < 16; ++i)
            a[i] = (a[i] > (_Float16)0) ? a[i] : (_Float16)0;
    }
    return a;
}

// 2 M-strips x 4 N-tiles per wave; each B-fragment feeds two WMMAs
__device__ __forceinline__ void gemm8(const _Float16* A, int strideA,
                                      const _Float16* __restrict__ W, int K,
                                      int n0, v8f acc[2][4], int lane, bool relu)
{
    const int ktc = K >> 5;
    for (int kt = 0; kt < ktc; ++kt) {
        v16h a0 = load_a_frag(A,                 strideA, kt << 5, lane, relu);
        v16h a1 = load_a_frag(A + 16 * strideA,  strideA, kt << 5, lane, relu);
        if (kt + 1 < ktc) {
            __builtin_prefetch(
                W + (((size_t)(n0 >> 4) * ktc + kt + 1) << 9) + lane * 16, 0, 3);
        }
        #pragma unroll
        for (int t = 0; t < 4; ++t) {
            const _Float16* bp =
                W + (((size_t)((n0 >> 4) + t) * ktc + kt) << 9) + lane * 16;
            v16h bfr = *(const v16h*)bp;
            acc[0][t] = __builtin_amdgcn_wmma_f32_16x16x32_f16(
                false, a0, false, bfr, (short)0, acc[0][t], false, false);
            acc[1][t] = __builtin_amdgcn_wmma_f32_16x16x32_f16(
                false, a1, false, bfr, (short)0, acc[1][t], false, false);
        }
    }
}

__global__ __launch_bounds__(256) void nerf_mlp_kernel(
    const _Float16* __restrict__ encW, const _Float16* __restrict__ algW,
    const _Float16* __restrict__ winP, const _Float16* __restrict__ blkP,
    const _Float16* __restrict__ woutP,
    const float* __restrict__ b_in, const float* __restrict__ b_z,
    const float* __restrict__ b_f0, const float* __restrict__ b_f1,
    const float* __restrict__ b_out, float* __restrict__ out)
{
    extern __shared__ __align__(16) _Float16 smem[];
    _Float16* sH = smem;                 // [32][DHP] running h
    _Float16* sT = smem + MT * DHP;      // [32][DHP] relu(net) temp
    _Float16* sZ = smem + 2 * MT * DHP;  // [32][DHP] aligned features
    _Float16* sE = smem + 3 * MT * DHP;  // [32][72]  encoding

    const int  tid  = threadIdx.x;
    const int  lane = tid & 31;
    const int  wave = tid >> 5;
    const long m0   = (long)blockIdx.x * MT;

    // stage enc tile: 32x64 halfs, 8 per thread
    {
        int r = tid >> 3, c = (tid & 7) * 8;
        *(v8h*)&sE[r * 72 + c] = *(const v8h*)(encW + (m0 + r) * DINP + c);
    }
    // stage aligned tile: 32x512 halfs, 64 per thread
    {
        int r = tid >> 3, c = (tid & 7) * 64;
        const _Float16* src = algW + (m0 + r) * DH + c;
        #pragma unroll
        for (int q = 0; q < 8; ++q)
            *(v8h*)&sZ[r * DHP + c + q * 8] = *(const v8h*)(src + q * 8);
    }
    __syncthreads();

    const int n0  = wave * 64;
    const int col = lane & 15;
    const int rb  = (lane < 16) ? 0 : 8;
    v8f acc[2][4];

    // ---- h = enc @ Win + b_in  (writes own columns only) -----------------
    #pragma unroll
    for (int s = 0; s < 2; ++s)
        #pragma unroll
        for (int t = 0; t < 4; ++t) acc[s][t] = vzero8();
    gemm8(sE, 72, winP, DINP, n0, acc, lane, false);
    #pragma unroll
    for (int s = 0; s < 2; ++s)
        #pragma unroll
        for (int t = 0; t < 4; ++t) {
            int n = n0 + t * 16 + col;
            float bv = b_in[n];
            #pragma unroll
            for (int e = 0; e < 8; ++e)
                sH[(s * 16 + rb + e) * DHP + n] = (_Float16)(acc[s][t][e] + bv);
        }

    // ---- 5 residual blocks (2 barriers per block) ------------------------
    for (int i = 0; i < NB; ++i) {
        // h += aligned @ Zw[i] + zb[i]
        #pragma unroll
        for (int s = 0; s < 2; ++s)
            #pragma unroll
            for (int t = 0; t < 4; ++t) acc[s][t] = vzero8();
        gemm8(sZ, DHP, blkP + (size_t)(i * 3 + 0) * BLK_MAT_HALFS,
              DH, n0, acc, lane, false);
        #pragma unroll
        for (int s = 0; s < 2; ++s)
            #pragma unroll
            for (int t = 0; t < 4; ++t) {
                int n = n0 + t * 16 + col;
                float bv = b_z[i * DH + n];
                #pragma unroll
                for (int e = 0; e < 8; ++e) {
                    int idx = (s * 16 + rb + e) * DHP + n;
                    sH[idx] = (_Float16)((float)sH[idx] + acc[s][t][e] + bv);
                }
            }
        __syncthreads();   // sH writes visible before all-column reads

        // net = relu(h) @ F0[i] + b0;  sT = relu(net)   (relu applied on load)
        #pragma unroll
        for (int s = 0; s < 2; ++s)
            #pragma unroll
            for (int t = 0; t < 4; ++t) acc[s][t] = vzero8();
        gemm8(sH, DHP, blkP + (size_t)(i * 3 + 1) * BLK_MAT_HALFS,
              DH, n0, acc, lane, true);
        #pragma unroll
        for (int s = 0; s < 2; ++s)
            #pragma unroll
            for (int t = 0; t < 4; ++t) {
                int n = n0 + t * 16 + col;
                float bv = b_f0[i * DH + n];
                #pragma unroll
                for (int e = 0; e < 8; ++e) {
                    float v = acc[s][t][e] + bv;
                    sT[(s * 16 + rb + e) * DHP + n] =
                        (_Float16)(v > 0.f ? v : 0.f);
                }
            }
        __syncthreads();   // sT writes visible before all-column reads

        // h += relu(net) @ F1[i] + b1
        #pragma unroll
        for (int s = 0; s < 2; ++s)
            #pragma unroll
            for (int t = 0; t < 4; ++t) acc[s][t] = vzero8();
        gemm8(sT, DHP, blkP + (size_t)(i * 3 + 2) * BLK_MAT_HALFS,
              DH, n0, acc, lane, false);
        #pragma unroll
        for (int s = 0; s < 2; ++s)
            #pragma unroll
            for (int t = 0; t < 4; ++t) {
                int n = n0 + t * 16 + col;
                float bv = b_f1[i * DH + n];
                #pragma unroll
                for (int e = 0; e < 8; ++e) {
                    int idx = (s * 16 + rb + e) * DHP + n;
                    sH[idx] = (_Float16)((float)sH[idx] + acc[s][t][e] + bv);
                }
            }
        // no barrier: next iteration's post-epi1 barrier orders sH writes
    }
    __syncthreads();

    // ---- sigma = exp(relu(h) @ Wout + b_out) -----------------------------
    if (wave < 2) {
        const _Float16* Abase = sH + wave * 16 * DHP;
        v8f c = vzero8();
        for (int kt = 0; kt < (DH >> 5); ++kt) {
            v16h a   = load_a_frag(Abase, DHP, kt << 5, lane, true);
            v16h bfr = *(const v16h*)(woutP + ((size_t)kt << 9) + lane * 16);
            c = __builtin_amdgcn_wmma_f32_16x16x32_f16(
                false, a, false, bfr, (short)0, c, false, false);
        }
        float bo = b_out[0];
        if (col == 0) {
            #pragma unroll
            for (int e = 0; e < 8; ++e)
                out[m0 + wave * 16 + rb + e] = expf(c[e] + bo);
        }
    }
}

// ---------------------------------------------------------------------------
// Host-side launcher
// ---------------------------------------------------------------------------
extern "C" void kernel_launch(void* const* d_in, const int* in_sizes, int n_in,
                              void* d_out, int out_size, void* d_ws, size_t ws_size,
                              hipStream_t stream)
{
    (void)in_sizes; (void)n_in; (void)out_size; (void)ws_size;

    const float* world_xyz = (const float*)d_in[0];
    const float* c2w       = (const float*)d_in[1];
    const float* k_mat     = (const float*)d_in[2];
    const float* features  = (const float*)d_in[3];
    const float* lin_in_w  = (const float*)d_in[4];
    const float* lin_in_b  = (const float*)d_in[5];
    const float* lin_z_w   = (const float*)d_in[6];
    const float* lin_z_b   = (const float*)d_in[7];
    const float* fc0_w     = (const float*)d_in[8];
    const float* fc0_b     = (const float*)d_in[9];
    const float* fc1_w     = (const float*)d_in[10];
    const float* fc1_b     = (const float*)d_in[11];
    const float* lin_out_w = (const float*)d_in[12];
    const float* lin_out_b = (const float*)d_in[13];
    float* out = (float*)d_out;

    const size_t M = M_TOTAL;
    _Float16* ws    = (_Float16*)d_ws;
    _Float16* encW  = ws;                                 // M*64
    _Float16* algW  = encW + M * DINP;                    // M*512
    _Float16* winP  = algW + M * DH;                      // 64 frags
    _Float16* blkP  = winP + (size_t)64 * FRAG_HALFS;     // 15*262144
    _Float16* woutP = blkP + (size_t)15 * BLK_MAT_HALFS;  // 16 frags

    nerf_prep_kernel<<<M_TOTAL, 256, 0, stream>>>(
        world_xyz, c2w, k_mat, features, encW, algW);

    const int NFRAG = 64 + 15 * FRAGS_512 + 16;
    nerf_pack_kernel<<<(NFRAG + 7) / 8, 256, 0, stream>>>(
        lin_in_w, lin_z_w, fc0_w, fc1_w, lin_out_w, winP, blkP, woutP);

    nerf_mlp_kernel<<<M_TOTAL / MT, 256, SMEM_BYTES, stream>>>(
        encW, algW, winP, blkP, woutP,
        lin_in_b, lin_z_b, fc0_b, fc1_b, lin_out_b, out);
}
